// EmbGCN_3470333575179
// MI455X (gfx1250) — matrix operand order repaired and checked by reading, not compiled, not run on hardware
//
#include <hip/hip_runtime.h>
#include <hip/hip_bf16.h>
#include <stdint.h>

// Problem constants (from reference)
#define BB 64
#define NN 2048
#define CI 32
#define CO 32
#define DD 16

typedef _Float16 v16h __attribute__((ext_vector_type(16)));
typedef _Float16 v8h  __attribute__((ext_vector_type(8)));
typedef float    v8f  __attribute__((ext_vector_type(8)));
typedef int      v4i  __attribute__((ext_vector_type(4)));

static __device__ __forceinline__ v8f wmma_f16(v16h a, v16h b, v8f c) {
    return __builtin_amdgcn_wmma_f32_16x16x32_f16(
        /*neg_a=*/false, a, /*neg_b=*/false, b,
        /*c_mod=*/(short)0, c, /*reuse_a=*/false, /*reuse_b=*/false);
}

// ---- CDNA5 async global->LDS copy (ASYNCcnt-tracked) with safe fallback ----
#if defined(__has_builtin)
#if __has_builtin(__builtin_amdgcn_global_load_async_to_lds_b128)
#define HAVE_ASYNC_LDS 1
#endif
#endif

#ifdef HAVE_ASYNC_LDS
typedef __attribute__((address_space(1))) v4i* as1_v4i_ptr;   // global (AS1)
typedef __attribute__((address_space(3))) v4i* as3_v4i_ptr;   // LDS (AS3)
#endif

static __device__ __forceinline__ void cp16_to_lds(const _Float16* g, _Float16* l) {
#ifdef HAVE_ASYNC_LDS
    // generic LDS pointer's low 32 bits == LDS byte offset (aperture rules)
    __builtin_amdgcn_global_load_async_to_lds_b128(
        (as1_v4i_ptr)(uintptr_t)g,
        (as3_v4i_ptr)(uint32_t)(uintptr_t)l,
        0, 0);
#else
    *(v8h*)l = *(const v8h*)g;   // sync load + ds_store fallback
#endif
}

static __device__ __forceinline__ void wait_async_pending2() {
#ifdef HAVE_ASYNC_LDS
    asm volatile("s_wait_asynccnt 0x2" ::: "memory");
#endif
}

// Fence-less workgroup split barrier: no VMEM/LDS counter drain (our protocol
// already orders LDS via s_wait_asynccnt before signal + ds data-deps).  The
// "memory" clobber only stops the COMPILER from moving memory ops across it.
static __device__ __forceinline__ void wg_barrier_relaxed() {
#ifdef HAVE_ASYNC_LDS
    asm volatile("s_barrier_signal -1\n\ts_barrier_wait -1" ::: "memory");
#else
    __syncthreads();   // fallback staging uses plain ds_store: need full fence
#endif
}

// ---------------------------------------------------------------------------
// Kernel 1: Ah[n][m] = softmax_m( relu(E[n] . E[m]) )  (f16 output)
// ---------------------------------------------------------------------------
__global__ void __launch_bounds__(256) k_adj(const float* __restrict__ E,
                                             _Float16* __restrict__ Ah) {
    const int row = blockIdx.x;
    const int t   = threadIdx.x;

    float er[DD];
#pragma unroll
    for (int d = 0; d < DD; ++d) er[d] = E[row * DD + d];

    float vals[8];
    float mx = 0.0f;  // relu output >= 0
#pragma unroll
    for (int j = 0; j < 8; ++j) {
        const int m = t + j * 256;
        const float4* em = (const float4*)(E + (size_t)m * DD);
        float4 a0 = em[0], a1 = em[1], a2 = em[2], a3 = em[3];
        float dot = er[0]*a0.x + er[1]*a0.y + er[2]*a0.z + er[3]*a0.w
                  + er[4]*a1.x + er[5]*a1.y + er[6]*a1.z + er[7]*a1.w
                  + er[8]*a2.x + er[9]*a2.y + er[10]*a2.z + er[11]*a2.w
                  + er[12]*a3.x + er[13]*a3.y + er[14]*a3.z + er[15]*a3.w;
        float r = dot > 0.0f ? dot : 0.0f;
        vals[j] = r;
        mx = fmaxf(mx, r);
    }

    __shared__ float red[256];
    red[t] = mx;
    __syncthreads();
    for (int s = 128; s > 0; s >>= 1) {
        if (t < s) red[t] = fmaxf(red[t], red[t + s]);
        __syncthreads();
    }
    mx = red[0];
    __syncthreads();

    float ex[8];
    float sum = 0.0f;
#pragma unroll
    for (int j = 0; j < 8; ++j) {
        ex[j] = __expf(vals[j] - mx);
        sum += ex[j];
    }
    red[t] = sum;
    __syncthreads();
    for (int s = 128; s > 0; s >>= 1) {
        if (t < s) red[t] += red[t + s];
        __syncthreads();
    }
    const float inv = 1.0f / red[0];
#pragma unroll
    for (int j = 0; j < 8; ++j) {
        Ah[(size_t)row * NN + t + j * 256] = (_Float16)(ex[j] * inv);
    }
}

// ---------------------------------------------------------------------------
// Kernel 2: WhT[n][o][kk] = sum_d E[n,d] * wp[d][kk/32][kk%32][o]   (f16)
//           biasN[n][o]   = sum_d E[n,d] * bp[d][o]                 (f32)
// ---------------------------------------------------------------------------
__global__ void __launch_bounds__(256) k_wt(const float* __restrict__ E,
                                            const float* __restrict__ wp,
                                            const float* __restrict__ bp,
                                            _Float16* __restrict__ WhT,
                                            float* __restrict__ biasN) {
    const int n = blockIdx.x;
    const int t = threadIdx.x;

    float en[DD];
#pragma unroll
    for (int d = 0; d < DD; ++d) en[d] = E[n * DD + d];

#pragma unroll
    for (int j = 0; j < 8; ++j) {
        const int f  = t + j * 256;     // f = o*64 + kk
        const int o  = f >> 6;
        const int kk = f & 63;
        float s = 0.0f;
#pragma unroll
        for (int d = 0; d < DD; ++d) s += en[d] * wp[d * 2048 + kk * 32 + o];
        WhT[(size_t)n * 2048 + f] = (_Float16)s;
    }
    if (t < CO) {
        float s = 0.0f;
#pragma unroll
        for (int d = 0; d < DD; ++d) s += en[d] * bp[d * CO + t];
        biasN[n * CO + t] = s;
    }
}

// ---------------------------------------------------------------------------
// Kernel 2b: XT[b][c][n] = (f16) x[b][n][c]
// ---------------------------------------------------------------------------
__global__ void __launch_bounds__(256) k_xt(const float* __restrict__ x,
                                            _Float16* __restrict__ XT) {
    const int wid  = (blockIdx.x * blockDim.x + threadIdx.x) >> 5;  // 0..4095
    const int lane = threadIdx.x & 31;
    const int b  = wid >> 6;
    const int n  = (wid & 63) * 32 + lane;

    const float4* px = (const float4*)(x + ((size_t)b * NN + n) * CI);
    float v[CI];
#pragma unroll
    for (int q = 0; q < 8; ++q) {
        float4 f = px[q];
        v[q * 4 + 0] = f.x; v[q * 4 + 1] = f.y;
        v[q * 4 + 2] = f.z; v[q * 4 + 3] = f.w;
    }
#pragma unroll
    for (int c = 0; c < CI; ++c) {
        XT[((size_t)b * CI + c) * NN + n] = (_Float16)v[c];
    }
}

// ---------------------------------------------------------------------------
// Kernel 3: Y[b] = Ah @ X[b].  Block = 8 waves x 16 rows (128 M-rows) sharing
// 4 batches.  B-operand staged in LDS (async, double-buffered, 40-half row
// stride => conflict-free ds_load_b128).  A-frags double-buffered in regs.
// ---------------------------------------------------------------------------
#define LROW 40                       // padded LDS row stride (halfs)

static __device__ __forceinline__ void agg_stage(const _Float16* __restrict__ g,
                                                 _Float16* __restrict__ l, int k0) {
    cp16_to_lds(g + k0,     l);
    cp16_to_lds(g + k0 + 8, l + 8);
}

static __device__ __forceinline__ void agg_compute(const _Float16* __restrict__ sbuf,
                                                   v16h a, int l16, int kB,
                                                   v8f acc[4][2]) {
#pragma unroll
    for (int g = 0; g < 4; ++g) {
        union { v16h v; v8h h[2]; } b0, b1;
        const _Float16* r0 = sbuf + (g * 32 + l16)      * LROW + kB;
        const _Float16* r1 = sbuf + (g * 32 + 16 + l16) * LROW + kB;
        b0.h[0] = *(const v8h*)(r0);  b0.h[1] = *(const v8h*)(r0 + 16);
        b1.h[0] = *(const v8h*)(r1);  b1.h[1] = *(const v8h*)(r1 + 16);
        acc[g][0] = wmma_f16(a, b0.v, acc[g][0]);
        acc[g][1] = wmma_f16(a, b1.v, acc[g][1]);
    }
}

__global__ void __launch_bounds__(256) k_agg(const _Float16* __restrict__ Ah,
                                             const _Float16* __restrict__ XT,
                                             _Float16* __restrict__ Y) {
    __shared__ _Float16 sB[2][128 * LROW];     // 2 x 10 KB

    const int t     = threadIdx.x;
    const int lane  = t & 31;
    const int w     = t >> 5;                  // wave in block: 0..7
    const int bg    = blockIdx.x >> 4;         // batch group: 0..15 (4 batches)
    const int rg    = blockIdx.x & 15;         // row group:   0..15 (128 rows)
    const int m0    = (rg * 8 + w) * 16;       // this wave's M-tile
    const int lhalf = lane >> 4;
    const int l16   = lane & 15;
    const int kA    = lhalf * 8;               // A frag K sub-offset
    const int kB    = lhalf * 8;               // B frag K sub-offset

    const _Float16* Abase = Ah + (size_t)(m0 + l16) * NN;

    // staging assignment: thread t copies halfs [sh, sh+16) of LDS row r
    const int r  = t >> 1;                     // 0..127  (= g*32 + col)
    const int sh = (t & 1) * 16;
    const _Float16* gstage = XT + ((size_t)(bg * 128 + r)) * NN + sh;
    _Float16* l0 = &sB[0][r * LROW + sh];
    _Float16* l1 = &sB[1][r * LROW + sh];

    v8f acc[4][2] = {};
    union AU { v16h v; v8h h[2]; } a_cur, a_nxt;

    agg_stage(gstage, l0, 0);
    a_cur.h[0] = *(const v8h*)(Abase + kA);
    a_cur.h[1] = *(const v8h*)(Abase + kA + 16);

    for (int kt = 0; kt < 64; kt += 2) {
        // ---- phase 0: compute buf0 (k=kt), prefetch buf1 (k=kt+1) ----
        const int k1 = (kt + 1) * 32;          // kt+1 <= 63 always
        agg_stage(gstage, l1, k1);
        a_nxt.h[0] = *(const v8h*)(Abase + k1 + kA);
        a_nxt.h[1] = *(const v8h*)(Abase + k1 + kA + 16);
        wait_async_pending2();                 // own buf0 writes complete
        wg_barrier_relaxed();                  // everyone's buf0 writes complete
        agg_compute(&sB[0][0], a_cur.v, l16, kB, acc);
        wg_barrier_relaxed();                  // everyone done reading buf0

        // ---- phase 1: compute buf1 (k=kt+1), prefetch buf0 (k=kt+2) ----
        const int k2 = (kt + 2 < 64) ? (kt + 2) * 32 : 63 * 32;  // clamped, uniform
        agg_stage(gstage, l0, k2);
        a_cur.h[0] = *(const v8h*)(Abase + k2 + kA);
        a_cur.h[1] = *(const v8h*)(Abase + k2 + kA + 16);
        wait_async_pending2();
        wg_barrier_relaxed();
        agg_compute(&sB[1][0], a_nxt.v, l16, kB, acc);
        wg_barrier_relaxed();
    }

#pragma unroll
    for (int g = 0; g < 4; ++g) {
        const int b = bg * 4 + g;
#pragma unroll
        for (int ct = 0; ct < 2; ++ct) {
            const int col = ct * 16 + l16;
#pragma unroll
            for (int v = 0; v < 8; ++v) {
                const int row = m0 + v + lhalf * 8;      // D layout: M = v + 8*lhalf
                Y[((size_t)b * NN + row) * CI + col] = (_Float16)acc[g][ct][v];
            }
        }
    }
}

// ---------------------------------------------------------------------------
// Kernel 4: out[b,n,o] = [x[b,n,:] | y[b,n,:]] @ WhT[n] + bias[n,o]
// Per node: M=64 (batches), K=64, N=32 -> 16 WMMAs. 2048 blocks x 1 wave.
// ---------------------------------------------------------------------------
__global__ void __launch_bounds__(32) k_conv(const float* __restrict__ x,
                                             const _Float16* __restrict__ Y,
                                             const _Float16* __restrict__ WhT,
                                             const float* __restrict__ biasN,
                                             float* __restrict__ out) {
    const int n     = blockIdx.x;
    const int lane  = threadIdx.x;
    const int lhalf = lane >> 4;
    const int l16   = lane & 15;
    const int ko    = lhalf * 16;
    const int kb    = lhalf * 16;

    v16h Bf[2][2];
#pragma unroll
    for (int ct = 0; ct < 2; ++ct) {
        const _Float16* wb = WhT + ((size_t)n * CO + ct * 16 + l16) * 64;
#pragma unroll
        for (int ks = 0; ks < 2; ++ks) {
            Bf[ks][ct] = *(const v16h*)(wb + ks * 32 + kb);
        }
    }
    const float bias0 = biasN[n * CO + l16];
    const float bias1 = biasN[n * CO + 16 + l16];

#pragma unroll
    for (int mt = 0; mt < 4; ++mt) {
        const int b = mt * 16 + l16;

        union { v16h v; v8h h[2]; } a0;
        const float* px = x + ((size_t)b * NN + n) * CI;
        v8f f0 = *(const v8f*)(px + ko);
        v8f f1 = *(const v8f*)(px + ko + 8);
        v8h h0, h1;
#pragma unroll
        for (int e = 0; e < 8; ++e) { h0[e] = (_Float16)f0[e]; h1[e] = (_Float16)f1[e]; }
        a0.h[0] = h0; a0.h[1] = h1;

        union { v16h v; v8h h[2]; } a1;
        const _Float16* py = Y + ((size_t)b * NN + n) * CI;
        a1.h[0] = *(const v8h*)(py + ko);
        a1.h[1] = *(const v8h*)(py + ko + 8);

        v8f acc0, acc1;
#pragma unroll
        for (int e = 0; e < 8; ++e) { acc0[e] = bias0; acc1[e] = bias1; }

        acc0 = wmma_f16(a0.v, Bf[0][0], acc0);
        acc0 = wmma_f16(a1.v, Bf[1][0], acc0);
        acc1 = wmma_f16(a0.v, Bf[0][1], acc1);
        acc1 = wmma_f16(a1.v, Bf[1][1], acc1);

#pragma unroll
        for (int v = 0; v < 8; ++v) {
            const int row = mt * 16 + v + lhalf * 8;
            out[((size_t)row * NN + n) * CO + l16]      = acc0[v];
            out[((size_t)row * NN + n) * CO + 16 + l16] = acc1[v];
        }
    }
}

// ---------------------------------------------------------------------------
extern "C" void kernel_launch(void* const* d_in, const int* in_sizes, int n_in,
                              void* d_out, int out_size, void* d_ws, size_t ws_size,
                              hipStream_t stream) {
    (void)in_sizes; (void)n_in; (void)out_size; (void)ws_size;

    const float* x  = (const float*)d_in[0];   // [B,N,CI]
    const float* E  = (const float*)d_in[1];   // [N,D]
    const float* wp = (const float*)d_in[2];   // [D,2,CI,CO]
    const float* bp = (const float*)d_in[3];   // [D,CO]
    float* out = (float*)d_out;                // [B,N,CO]

    char* ws = (char*)d_ws;
    _Float16* Ah    = (_Float16*)(ws);                              // 8 MB  [N,N]
    _Float16* XT    = (_Float16*)(ws + (8u  << 20));                // 8 MB  [B,CI,N]
    _Float16* WhT   = (_Float16*)(ws + (16u << 20));                // 8 MB  [N,CO,64]
    float*    biasN = (float*)   (ws + (24u << 20));                // 256KB [N,CO]
    _Float16* Yb    = (_Float16*)(ws + (24u << 20) + (256u << 10)); // 8 MB  [B,N,CI]

    k_adj <<<NN,                 256, 0, stream>>>(E, Ah);
    k_wt  <<<NN,                 256, 0, stream>>>(E, wp, bp, WhT, biasN);
    k_xt  <<<(BB * (NN/32)) / 8, 256, 0, stream>>>(x, XT);
    k_agg <<<16 * 16,            256, 0, stream>>>(Ah, XT, Yb);
    k_conv<<<NN,                 32,  0, stream>>>(x, Yb, WhT, biasN, out);
}
